// Model_60868276518949
// MI455X (gfx1250) — matrix operand order repaired and checked by reading, not compiled, not run on hardware
//
#include <hip/hip_runtime.h>
#include <hip/hip_bf16.h>
#include <math.h>

// ---------------- model constants ----------------
#define B_    8
#define T_    16
#define H_    28
#define W_    28
#define D1_   192
#define D2_   128
#define SEQ_N 12544              // T*H*W tokens per batch
#define VOXN  100352             // B * SEQ_N total tokens
#define CHUNK_M 14336            // VOXN / 7, multiple of 128

static const size_t ND1 = (size_t)VOXN * D1_;   // 19,267,584
static const size_t ND2 = (size_t)VOXN * D2_;   // 12,845,056

// ---------------- WMMA vector types ----------------
typedef __attribute__((ext_vector_type(16))) _Float16 v16h;
typedef __attribute__((ext_vector_type(8)))  float    v8f;

__device__ __forceinline__ float geluf(float x) {
    // branch-free tanh-GELU: tanh(u) = sign(u) * (1 - e) / (1 + e), e = exp(-2|u|)
    const float k0 = 0.7978845608028654f;  // sqrt(2/pi)
    float u = k0 * (x + 0.044715f * x * x * x);
    float e = __expf(-2.0f * fabsf(u));
    float th = __builtin_copysignf((1.0f - e) / (1.0f + e), u);
    return 0.5f * x * (1.0f + th);
}

// =====================================================================
// Pack fp32 weight W[K,N] -> f16 in WMMA B-operand tile layout:
//   P[((kt*(N/16) + nt)*32 + lane)*16 + e] = W[(kt*32 + (lane>>4)*8 + e + (e>>3)*8)*N
//                                              + nt*16 + (lane&15)]
// =====================================================================
__global__ __launch_bounds__(256)
void pack_w(const float* __restrict__ W, _Float16* __restrict__ P, int K, int N)
{
    int idx = blockIdx.x * 256 + threadIdx.x;
    if (idx >= K * N) return;
    int e    = idx & 15;
    int l    = (idx >> 4) & 31;
    int tile = idx >> 9;
    int ntiles = N >> 4;
    int nt = tile % ntiles;
    int kt = tile / ntiles;
    int ln = l & 15, hi = l >> 4;
    int k  = kt * 32 + hi * 8 + e + ((e >> 3) << 3);
    P[idx] = (_Float16)W[(size_t)k * N + nt * 16 + ln];
}

// =====================================================================
// Layout transpose: x (B,D1,T,H,W) -> xf (B,T,H,W,D1)
// =====================================================================
__global__ __launch_bounds__(256)
void transpose_in(const float* __restrict__ x, float* __restrict__ xf)
{
    size_t idx = (size_t)blockIdx.x * 256 + threadIdx.x;
    if (idx >= (size_t)VOXN * D1_) return;
    int c = (int)(idx % D1_);
    size_t t = idx / D1_;
    int w = (int)(t % W_); t /= W_;
    int h = (int)(t % H_); t /= H_;
    int tt = (int)(t % T_); int b = (int)(t / T_);
    xf[idx] = x[((((size_t)b * D1_ + c) * T_ + tt) * H_ + h) * W_ + w];
}

// =====================================================================
// Per-row LayerNorm: one block per row, C = 192 or 128
// =====================================================================
__global__ __launch_bounds__(256)
void ln_kernel(const float* __restrict__ x, const float* __restrict__ g,
               const float* __restrict__ bb, float* __restrict__ out, int C)
{
    __shared__ float red[256];
    const size_t row = blockIdx.x;
    const float* xr = x + row * (size_t)C;
    const int tid = threadIdx.x;
    float s = 0.f;
    for (int c = tid; c < C; c += 256) s += xr[c];
    red[tid] = s; __syncthreads();
    for (int off = 128; off > 0; off >>= 1) {
        if (tid < off) red[tid] += red[tid + off];
        __syncthreads();
    }
    const float mean = red[0] * (1.0f / C);
    __syncthreads();
    float v = 0.f;
    for (int c = tid; c < C; c += 256) { float d = xr[c] - mean; v += d * d; }
    red[tid] = v; __syncthreads();
    for (int off = 128; off > 0; off >>= 1) {
        if (tid < off) red[tid] += red[tid + off];
        __syncthreads();
    }
    const float inv = rsqrtf(red[0] * (1.0f / C) + 1e-5f);
    float* orow = out + row * (size_t)C;
    for (int c = tid; c < C; c += 256)
        orow[c] = (xr[c] - mean) * inv * g[c] + bb[c];
}

// =====================================================================
// Depthwise 3x3x3 conv, SAME padding, NDHWC, groups = D1
// =====================================================================
__global__ __launch_bounds__(256)
void dwconv(const float* __restrict__ hb, const float* __restrict__ wgt,
            float* __restrict__ out)
{
    size_t idx = (size_t)blockIdx.x * 256 + threadIdx.x;
    if (idx >= (size_t)VOXN * D1_) return;
    int c = (int)(idx % D1_);
    size_t t = idx / D1_;
    int w = (int)(t % W_); t /= W_;
    int h = (int)(t % H_); t /= H_;
    int tt = (int)(t % T_); int b = (int)(t / T_);
    float acc = 0.f;
    for (int dt = 0; dt < 3; ++dt) {
        int t2 = tt + dt - 1; if (t2 < 0 || t2 >= T_) continue;
        for (int dh = 0; dh < 3; ++dh) {
            int h2 = h + dh - 1; if (h2 < 0 || h2 >= H_) continue;
            for (int dw = 0; dw < 3; ++dw) {
                int w2 = w + dw - 1; if (w2 < 0 || w2 >= W_) continue;
                acc += hb[((((size_t)b * T_ + t2) * H_ + h2) * W_ + w2) * D1_ + c]
                     * wgt[((dt * 3 + dh) * 3 + dw) * D1_ + c];
            }
        }
    }
    out[idx] = acc;
}

// =====================================================================
// WMMA GEMM: out[M,N] = epilogue(A[M,K] @ W[K,N] + bias[N]) (+ res)
//   Wp = f16 weights pre-packed in B-operand tile layout.
//   block = 256 threads (8 waves); wave w owns rows [blockIdx.y*128+16w, +16)
//   and a 64-wide column strip (4 accumulators) at blockIdx.x*64.
//   K mult of 32, N mult of 64, M mult of 128. act: 0=none, 1=GELU.
// =====================================================================
__global__ __launch_bounds__(256)
void wmma_gemm(const float* __restrict__ A, const _Float16* __restrict__ Wp,
               const float* __restrict__ bias, const float* res,
               float* out, int N, int K, int act)
{
    const int wave = threadIdx.x >> 5;
    const int lane = threadIdx.x & 31;
    const int ln   = lane & 15;
    const int hi   = lane >> 4;
    const int n0   = blockIdx.x * 64;
    const int m0   = blockIdx.y * 128 + wave * 16;
    const int ntiles = N >> 4;

    v8f acc0 = {0.f,0.f,0.f,0.f,0.f,0.f,0.f,0.f};
    v8f acc1 = acc0, acc2 = acc0, acc3 = acc0;

    // A operand: lane holds row (m0+ln); K runs kBase+0..7 and kBase+16..23
    const float* Arow = A + (size_t)(m0 + ln) * K + hi * 8;
    for (int kk = 0; kk < K; kk += 32) {
        __builtin_prefetch(Arow + kk + 64, 0, 1);
        const float4* ap = (const float4*)(Arow + kk);
        float4 a0 = ap[0], a1 = ap[1], a2 = ap[4], a3 = ap[5];
        v16h av;
        av[0]  = (_Float16)a0.x; av[1]  = (_Float16)a0.y;
        av[2]  = (_Float16)a0.z; av[3]  = (_Float16)a0.w;
        av[4]  = (_Float16)a1.x; av[5]  = (_Float16)a1.y;
        av[6]  = (_Float16)a1.z; av[7]  = (_Float16)a1.w;
        av[8]  = (_Float16)a2.x; av[9]  = (_Float16)a2.y;
        av[10] = (_Float16)a2.z; av[11] = (_Float16)a2.w;
        av[12] = (_Float16)a3.x; av[13] = (_Float16)a3.y;
        av[14] = (_Float16)a3.z; av[15] = (_Float16)a3.w;
        // B operands: 16 contiguous halves per lane per tile
        const _Float16* bp = Wp + ((size_t)(kk >> 5) * ntiles + (n0 >> 4)) * 512
                                + lane * 16;
        v16h b0 = *(const v16h*)(bp);
        v16h b1 = *(const v16h*)(bp + 512);
        v16h b2 = *(const v16h*)(bp + 1024);
        v16h b3 = *(const v16h*)(bp + 1536);
        acc0 = __builtin_amdgcn_wmma_f32_16x16x32_f16(false, av, false, b0, (short)0, acc0, false, false);
        acc1 = __builtin_amdgcn_wmma_f32_16x16x32_f16(false, av, false, b1, (short)0, acc1, false, false);
        acc2 = __builtin_amdgcn_wmma_f32_16x16x32_f16(false, av, false, b2, (short)0, acc2, false, false);
        acc3 = __builtin_amdgcn_wmma_f32_16x16x32_f16(false, av, false, b3, (short)0, acc3, false, false);
    }
    #pragma unroll
    for (int t = 0; t < 4; ++t) {
        v8f acc = (t == 0) ? acc0 : (t == 1) ? acc1 : (t == 2) ? acc2 : acc3;
        const int nb = n0 + t * 16 + ln;
        #pragma unroll
        for (int r = 0; r < 8; ++r) {
            const int m = m0 + hi * 8 + r;
            float vv = acc[r] + bias[nb];
            if (act) vv = geluf(vv);
            if (res) vv += res[(size_t)m * N + nb];
            out[(size_t)m * N + nb] = vv;
        }
    }
}

// =====================================================================
// Local windowed attention, heads 0..7, WIN=16, DH=8. One wave per
// (window, head, batch). Scores + PV both via zero-padded f16 WMMA.
// =====================================================================
__global__ __launch_bounds__(32)
void local_attn(const float* __restrict__ q, const float* __restrict__ k,
                const float* __restrict__ v, float* __restrict__ o)
{
    __shared__ float sq[16][8], sk[16][8], sv[16][8], sp[16][16];
    const int win = blockIdx.x, head = blockIdx.y, b = blockIdx.z;
    const int lane = threadIdx.x;
    const int ln = lane & 15, hi = lane >> 4;
    const size_t tok0 = (size_t)b * SEQ_N + (size_t)win * 16;

    for (int t = 0; t < 4; ++t) {
        int idx = lane * 4 + t;          // 0..127 -> (i, d)
        int i = idx >> 3, d = idx & 7;
        size_t off = (tok0 + i) * D2_ + head * 8 + d;
        sq[i][d] = q[off] * 0.35355339059327373f;   // fold DH^-0.5
        sk[i][d] = k[off];
        sv[i][d] = v[off];
    }
    __syncthreads();

    v16h av, bv;
    #pragma unroll
    for (int e = 0; e < 16; ++e) {
        int ka = hi * 8 + e + ((e >> 3) << 3);   // A layout K index
        int kb = hi * 8 + e + ((e >> 3) << 3);   // B layout K index
        av[e] = (ka < 8) ? (_Float16)sq[ln][ka] : (_Float16)0.f;
        bv[e] = (kb < 8) ? (_Float16)sk[ln][kb] : (_Float16)0.f;
    }
    v8f s = {0.f,0.f,0.f,0.f,0.f,0.f,0.f,0.f};
    s = __builtin_amdgcn_wmma_f32_16x16x32_f16(
            false, av, false, bv, (short)0, s, false, false);

    // row softmax: row (hi*8 + r) lives in VGPR r across the 16 lanes of a half
    #pragma unroll
    for (int r = 0; r < 8; ++r) {
        float val = s[r];
        float mx = val;
        for (int msk = 1; msk < 16; msk <<= 1) mx = fmaxf(mx, __shfl_xor(mx, msk));
        float p = __expf(val - mx);
        float sum = p;
        for (int msk = 1; msk < 16; msk <<= 1) sum += __shfl_xor(sum, msk);
        sp[hi * 8 + r][ln] = p / sum;
    }
    __syncthreads();

    #pragma unroll
    for (int e = 0; e < 16; ++e) {
        int ka = hi * 8 + e + ((e >> 3) << 3);
        int kb = hi * 8 + e + ((e >> 3) << 3);
        av[e] = (ka < 16) ? (_Float16)sp[ln][ka] : (_Float16)0.f;
        bv[e] = (kb < 16 && ln < 8) ? (_Float16)sv[kb][ln] : (_Float16)0.f;
    }
    v8f oa = {0.f,0.f,0.f,0.f,0.f,0.f,0.f,0.f};
    oa = __builtin_amdgcn_wmma_f32_16x16x32_f16(
             false, av, false, bv, (short)0, oa, false, false);

    if (ln < 8) {
        #pragma unroll
        for (int r = 0; r < 8; ++r)
            o[(tok0 + hi * 8 + r) * D2_ + head * 8 + ln] = oa[r];
    }
}

// =====================================================================
// Performer (heads 8..15): key-side global max of dash over (n, m)
// =====================================================================
__global__ __launch_bounds__(256)
void perf_kmax(const float* __restrict__ k, const float* __restrict__ proj,
               float* __restrict__ kmax)
{
    __shared__ float sproj[16][8];
    __shared__ float red[256];
    const int h = blockIdx.x, b = blockIdx.y;
    const int tid = threadIdx.x;
    if (tid < 128) sproj[tid >> 3][tid & 7] = proj[tid];
    __syncthreads();
    float mx = -3.0e38f;
    for (int n = tid; n < SEQ_N; n += 256) {
        const float* kn = k + ((size_t)b * SEQ_N + n) * D2_ + (8 + h) * 8;
        float xn[8];
        #pragma unroll
        for (int d = 0; d < 8; ++d) xn[d] = kn[d] * 0.5946035575013605f; // DH^-0.25
        #pragma unroll
        for (int m = 0; m < 16; ++m) {
            float dd = 0.f;
            #pragma unroll
            for (int d = 0; d < 8; ++d) dd += xn[d] * sproj[m][d];
            mx = fmaxf(mx, dd);
        }
    }
    red[tid] = mx; __syncthreads();
    for (int off = 128; off > 0; off >>= 1) {
        if (tid < off) red[tid] = fmaxf(red[tid], red[tid + off]);
        __syncthreads();
    }
    if (tid == 0) kmax[b * 8 + h] = red[0];
}

// kp[b,h,n,m] = NB^-0.5 * (exp(dash - diag - kmax) + 1e-6)
__global__ __launch_bounds__(256)
void perf_kp(const float* __restrict__ k, const float* __restrict__ proj,
             const float* __restrict__ kmax, float* __restrict__ kp)
{
    __shared__ float sproj[16][8];
    const int h = blockIdx.y, b = blockIdx.z;
    const int tid = threadIdx.x;
    if (tid < 128) sproj[tid >> 3][tid & 7] = proj[tid];
    __syncthreads();
    const int n = blockIdx.x * 256 + tid;
    const int bh = b * 8 + h;
    const float km = kmax[bh];
    const float* kn = k + ((size_t)b * SEQ_N + n) * D2_ + (8 + h) * 8;
    float xn[8]; float diag = 0.f;
    #pragma unroll
    for (int d = 0; d < 8; ++d) { xn[d] = kn[d] * 0.5946035575013605f; diag += xn[d] * xn[d]; }
    diag *= 0.5f;
    float* kpn = kp + ((size_t)bh * SEQ_N + n) * 16;
    #pragma unroll
    for (int m = 0; m < 16; ++m) {
        float dd = 0.f;
        #pragma unroll
        for (int d = 0; d < 8; ++d) dd += xn[d] * sproj[m][d];
        kpn[m] = 0.25f * (__expf(dd - diag - km) + 1e-6f);
    }
}

// ctx[b,h,m,d] = sum_n kp * v ; ksum[b,h,m] = sum_n kp
__global__ __launch_bounds__(256)
void perf_ctx(const float* __restrict__ kp, const float* __restrict__ v,
              float* __restrict__ ctx, float* __restrict__ ksum)
{
    const int h = blockIdx.x, b = blockIdx.y;
    const int bh = b * 8 + h;
    const int tid = threadIdx.x;
    if (tid < 128) {
        const int m = tid >> 3, d = tid & 7;
        float acc = 0.f;
        for (int n = 0; n < SEQ_N; ++n)
            acc += kp[((size_t)bh * SEQ_N + n) * 16 + m]
                 * v[((size_t)b * SEQ_N + n) * D2_ + (8 + h) * 8 + d];
        ctx[bh * 128 + tid] = acc;
    } else if (tid < 144) {
        const int m = tid - 128;
        float acc = 0.f;
        for (int n = 0; n < SEQ_N; ++n)
            acc += kp[((size_t)bh * SEQ_N + n) * 16 + m];
        ksum[bh * 16 + m] = acc;
    }
}

// og[n,d] = (qp @ ctx) / (qp . ksum), qp computed on the fly (per-row max)
__global__ __launch_bounds__(256)
void perf_og(const float* __restrict__ q, const float* __restrict__ proj,
             const float* __restrict__ ctx, const float* __restrict__ ksum,
             float* __restrict__ o)
{
    __shared__ float sproj[16][8];
    __shared__ float sctx[16][8];
    __shared__ float sksum[16];
    const int h = blockIdx.y, b = blockIdx.z;
    const int bh = b * 8 + h;
    const int tid = threadIdx.x;
    if (tid < 128) {
        sproj[tid >> 3][tid & 7] = proj[tid];
        sctx[tid >> 3][tid & 7]  = ctx[bh * 128 + tid];
    }
    if (tid < 16) sksum[tid] = ksum[bh * 16 + tid];
    __syncthreads();
    const int n = blockIdx.x * 256 + tid;
    const float* qn = q + ((size_t)b * SEQ_N + n) * D2_ + (8 + h) * 8;
    float xn[8]; float diag = 0.f;
    #pragma unroll
    for (int d = 0; d < 8; ++d) { xn[d] = qn[d] * 0.5946035575013605f; diag += xn[d] * xn[d]; }
    diag *= 0.5f;
    float dash[16]; float mx = -3.0e38f;
    #pragma unroll
    for (int m = 0; m < 16; ++m) {
        float dd = 0.f;
        #pragma unroll
        for (int d = 0; d < 8; ++d) dd += xn[d] * sproj[m][d];
        dash[m] = dd; mx = fmaxf(mx, dd);
    }
    float num[8] = {0,0,0,0,0,0,0,0}; float den = 0.f;
    #pragma unroll
    for (int m = 0; m < 16; ++m) {
        float qp = 0.25f * (__expf(dash[m] - diag - mx) + 1e-6f);
        den += qp * sksum[m];
        #pragma unroll
        for (int d = 0; d < 8; ++d) num[d] += qp * sctx[m][d];
    }
    const float inv = 1.f / den;
    float* on = o + ((size_t)b * SEQ_N + n) * D2_ + (8 + h) * 8;
    #pragma unroll
    for (int d = 0; d < 8; ++d) on[d] = num[d] * inv;
}

// =====================================================================
// Final: feat = LN(max_n seq), logits = feat @ fc_w + fc_b
// d_out layout: [logits (8)] [feat (8*128)]
// =====================================================================
__global__ __launch_bounds__(128)
void final_head(const float* __restrict__ seq, const float* __restrict__ g,
                const float* __restrict__ bta, const float* __restrict__ fcw,
                const float* __restrict__ fcb, float* __restrict__ out)
{
    __shared__ float red[128];
    const int b = blockIdx.x, c = threadIdx.x;
    float mx = -3.0e38f;
    for (int n = 0; n < SEQ_N; ++n)
        mx = fmaxf(mx, seq[((size_t)b * SEQ_N + n) * D2_ + c]);
    red[c] = mx; __syncthreads();
    for (int off = 64; off > 0; off >>= 1) { if (c < off) red[c] += red[c + off]; __syncthreads(); }
    const float mean = red[0] * (1.f / 128.f);
    __syncthreads();
    const float dv = mx - mean;
    red[c] = dv * dv; __syncthreads();
    for (int off = 64; off > 0; off >>= 1) { if (c < off) red[c] += red[c + off]; __syncthreads(); }
    const float inv = rsqrtf(red[0] * (1.f / 128.f) + 1e-5f);
    __syncthreads();
    const float feat = dv * inv * g[c] + bta[c];
    out[8 + b * 128 + c] = feat;
    red[c] = feat * fcw[c]; __syncthreads();
    for (int off = 64; off > 0; off >>= 1) { if (c < off) red[c] += red[c + off]; __syncthreads(); }
    if (c == 0) out[b] = red[0] + fcb[0];
}

// =====================================================================
// Host orchestration
// =====================================================================
extern "C" void kernel_launch(void* const* d_in, const int* in_sizes, int n_in,
                              void* d_out, int out_size, void* d_ws, size_t ws_size,
                              hipStream_t stream)
{
    (void)in_sizes; (void)n_in; (void)out_size; (void)ws_size;

    const float* x_in     = (const float*)d_in[0];
    const float* cb_ln1_g = (const float*)d_in[1];
    const float* cb_ln1_b = (const float*)d_in[2];
    const float* cb_dw    = (const float*)d_in[3];
    const float* cb_pw_w  = (const float*)d_in[4];
    const float* cb_pw_b  = (const float*)d_in[5];
    const float* cb_ln2_g = (const float*)d_in[6];
    const float* cb_ln2_b = (const float*)d_in[7];
    const float* cb_ff_w1 = (const float*)d_in[8];
    const float* cb_ff_b1 = (const float*)d_in[9];
    const float* cb_ff_w2 = (const float*)d_in[10];
    const float* cb_ff_b2 = (const float*)d_in[11];
    const float* tr_ln_g  = (const float*)d_in[12];
    const float* tr_ln_b  = (const float*)d_in[13];
    const float* tr_w     = (const float*)d_in[14];
    const float* tr_b     = (const float*)d_in[15];
    const float* ab_ln1_g = (const float*)d_in[16];
    const float* ab_ln1_b = (const float*)d_in[17];
    const float* ab_wq    = (const float*)d_in[18];
    const float* ab_wk    = (const float*)d_in[19];
    const float* ab_wv    = (const float*)d_in[20];
    const float* ab_wo    = (const float*)d_in[21];
    const float* ab_bq    = (const float*)d_in[22];
    const float* ab_bk    = (const float*)d_in[23];
    const float* ab_bv    = (const float*)d_in[24];
    const float* ab_bo    = (const float*)d_in[25];
    const float* ab_proj  = (const float*)d_in[26];
    const float* ab_ln2_g = (const float*)d_in[27];
    const float* ab_ln2_b = (const float*)d_in[28];
    const float* ab_ff_w1 = (const float*)d_in[29];
    const float* ab_ff_b1 = (const float*)d_in[30];
    const float* ab_ff_w2 = (const float*)d_in[31];
    const float* ab_ff_b2 = (const float*)d_in[32];
    const float* fin_ln_g = (const float*)d_in[33];
    const float* fin_ln_b = (const float*)d_in[34];
    const float* fc_w     = (const float*)d_in[35];
    const float* fc_b     = (const float*)d_in[36];

    // -------- workspace layout (floats) --------
    float* ws   = (float*)d_ws;
    float* w_x  = ws;                 // xf / seq          (ND1)
    float* w_h  = w_x + ND1;          // LN outputs        (ND1)
    float* w_c  = w_h + ND1;          // dwconv out        (ND1)
    float* w_q  = w_c + ND1;          // q                 (ND2)
    float* w_k  = w_q + ND2;          // k                 (ND2)
    float* w_v  = w_k + ND2;          // v                 (ND2)
    float* w_o  = w_v + ND2;          // attn out          (ND2)
    float* w_kp = w_o + ND2;          // performer kp      (ND2)
    float* w_km = w_kp + ND2;         // kmax              (64)
    float* w_cx = w_km + 64;          // ctx               (64*128)
    float* w_ks = w_cx + 64 * 128;    // ksum              (64*16)
    float* w_t  = w_ks + 64 * 16;     // FFN chunk         (CHUNK_M*768)

    // -------- packed f16 weights (appended after w_t, 32B-aligned) --------
    _Float16* pcur = (_Float16*)(w_t + (size_t)CHUNK_M * 768);
    auto pack = [&](const float* W, int K, int N) -> _Float16* {
        _Float16* dst = pcur;
        int total = K * N;
        pack_w<<<(total + 255) / 256, 256, 0, stream>>>(W, dst, K, N);
        pcur += total;
        return dst;
    };
    _Float16 *p_pw[3], *p_f1[3], *p_f2[3], *p_wq[3], *p_wk[3], *p_wv[3],
             *p_wo[3], *p_a1[3], *p_a2[3];
    for (int i = 0; i < 3; ++i) {
        p_pw[i] = pack(cb_pw_w + (size_t)i * D1_ * D1_, D1_, D1_);
        p_f1[i] = pack(cb_ff_w1 + (size_t)i * D1_ * 768, D1_, 768);
        p_f2[i] = pack(cb_ff_w2 + (size_t)i * 768 * D1_, 768, D1_);
    }
    _Float16* p_tr = pack(tr_w, D1_, D2_);
    for (int i = 0; i < 3; ++i) {
        p_wq[i] = pack(ab_wq + (size_t)i * D2_ * D2_, D2_, D2_);
        p_wk[i] = pack(ab_wk + (size_t)i * D2_ * D2_, D2_, D2_);
        p_wv[i] = pack(ab_wv + (size_t)i * D2_ * D2_, D2_, D2_);
        p_wo[i] = pack(ab_wo + (size_t)i * D2_ * D2_, D2_, D2_);
        p_a1[i] = pack(ab_ff_w1 + (size_t)i * D2_ * 512, D2_, 512);
        p_a2[i] = pack(ab_ff_w2 + (size_t)i * 512 * D2_, 512, D2_);
    }

    const int ln_blocks = VOXN;
    const int el_blocks = (int)(ND1 / 256);     // 75264 exactly

    // -------- input transpose --------
    transpose_in<<<el_blocks, 256, 0, stream>>>(x_in, w_x);

    // -------- conv blocks --------
    for (int i = 0; i < 3; ++i) {
        ln_kernel<<<ln_blocks, 256, 0, stream>>>(w_x, cb_ln1_g + i * D1_, cb_ln1_b + i * D1_, w_h, D1_);
        dwconv<<<el_blocks, 256, 0, stream>>>(w_h, cb_dw + (size_t)i * 27 * D1_, w_c);
        wmma_gemm<<<dim3(D1_ / 64, VOXN / 128), 256, 0, stream>>>(
            w_c, p_pw[i], cb_pw_b + i * D1_, w_x, w_x, D1_, D1_, 0);
        ln_kernel<<<ln_blocks, 256, 0, stream>>>(w_x, cb_ln2_g + i * D1_, cb_ln2_b + i * D1_, w_h, D1_);
        for (int c0 = 0; c0 < VOXN; c0 += CHUNK_M) {
            wmma_gemm<<<dim3(768 / 64, CHUNK_M / 128), 256, 0, stream>>>(
                w_h + (size_t)c0 * D1_, p_f1[i], cb_ff_b1 + i * 768,
                nullptr, w_t, 768, D1_, 1);
            wmma_gemm<<<dim3(D1_ / 64, CHUNK_M / 128), 256, 0, stream>>>(
                w_t, p_f2[i], cb_ff_b2 + i * D1_,
                w_x + (size_t)c0 * D1_, w_x + (size_t)c0 * D1_, D1_, 768, 0);
        }
    }

    // -------- transition to D2, seq lives in w_x [VOXN, 128] --------
    ln_kernel<<<ln_blocks, 256, 0, stream>>>(w_x, tr_ln_g, tr_ln_b, w_h, D1_);
    wmma_gemm<<<dim3(D2_ / 64, VOXN / 128), 256, 0, stream>>>(
        w_h, p_tr, tr_b, nullptr, w_x, D2_, D1_, 0);

    // -------- attention blocks --------
    for (int i = 0; i < 3; ++i) {
        const float* pr = ab_proj + (size_t)i * 16 * 8;

        ln_kernel<<<ln_blocks, 256, 0, stream>>>(w_x, ab_ln1_g + i * D2_, ab_ln1_b + i * D2_, w_h, D2_);
        wmma_gemm<<<dim3(D2_ / 64, VOXN / 128), 256, 0, stream>>>(
            w_h, p_wq[i], ab_bq + i * D2_, nullptr, w_q, D2_, D2_, 0);
        wmma_gemm<<<dim3(D2_ / 64, VOXN / 128), 256, 0, stream>>>(
            w_h, p_wk[i], ab_bk + i * D2_, nullptr, w_k, D2_, D2_, 0);
        wmma_gemm<<<dim3(D2_ / 64, VOXN / 128), 256, 0, stream>>>(
            w_h, p_wv[i], ab_bv + i * D2_, nullptr, w_v, D2_, D2_, 0);

        local_attn<<<dim3(SEQ_N / 16, 8, B_), 32, 0, stream>>>(w_q, w_k, w_v, w_o);

        perf_kmax<<<dim3(8, B_), 256, 0, stream>>>(w_k, pr, w_km);
        perf_kp<<<dim3(SEQ_N / 256, 8, B_), 256, 0, stream>>>(w_k, pr, w_km, w_kp);
        perf_ctx<<<dim3(8, B_), 256, 0, stream>>>(w_kp, w_v, w_cx, w_ks);
        perf_og<<<dim3(SEQ_N / 256, 8, B_), 256, 0, stream>>>(w_q, pr, w_cx, w_ks, w_o);

        wmma_gemm<<<dim3(D2_ / 64, VOXN / 128), 256, 0, stream>>>(
            w_o, p_wo[i], ab_bo + i * D2_, w_x, w_x, D2_, D2_, 0);

        ln_kernel<<<ln_blocks, 256, 0, stream>>>(w_x, ab_ln2_g + i * D2_, ab_ln2_b + i * D2_, w_h, D2_);
        for (int c0 = 0; c0 < VOXN; c0 += CHUNK_M) {
            wmma_gemm<<<dim3(512 / 64, CHUNK_M / 128), 256, 0, stream>>>(
                w_h + (size_t)c0 * D2_, p_a1[i], ab_ff_b1 + i * 512,
                nullptr, w_t, 512, D2_, 1);
            wmma_gemm<<<dim3(D2_ / 64, CHUNK_M / 128), 256, 0, stream>>>(
                w_t, p_a2[i], ab_ff_b2 + i * D2_,
                w_x + (size_t)c0 * D2_, w_x + (size_t)c0 * D2_, D2_, 512, 0);
        }
    }

    // -------- final head --------
    final_head<<<B_, 128, 0, stream>>>(w_x, fin_ln_g, fin_ln_b, fc_w, fc_b, (float*)d_out);
}
